// WindowMSA_26199300505804
// MI455X (gfx1250) — matrix operand order repaired and checked by reading, not compiled, not run on hardware
//
#include <hip/hip_runtime.h>
#include <hip/hip_bf16.h>

typedef __attribute__((ext_vector_type(16))) _Float16 v16h;
typedef __attribute__((ext_vector_type(8)))  _Float16 v8h;
typedef __attribute__((ext_vector_type(8)))  float    v8f;

#define C_DIM   192
#define N_TOK   49
#define N_PAD   64
#define HEADS   6
#define HD      32
#define NWIN    64

// ---------------------------------------------------------------------------
// Fragment loaders following the CDNA5 wave32 WMMA VGPR layouts (ISA 7.12.2).
//
// A (16x32 f16, M x K): lane l holds row m=l&15. halves 0..7 -> K = off8 + j,
// halves 8..15 -> K = 16 + off8 + (j-8), off8 = (l&16)?8:0. Two contiguous
// 8-half chunks from a row-major [m][k] tile.
// ---------------------------------------------------------------------------
__device__ __forceinline__ v16h load_a_frag(const _Float16* base, int ld, int k0) {
  int lane = threadIdx.x & 31;
  int m    = lane & 15;
  int off  = (lane & 16) ? 8 : 0;
  const _Float16* p = base + m * ld + k0 + off;
  v8h lo = *(const v8h*)(p);
  v8h hi = *(const v8h*)(p + 16);
  v16h a;
#pragma unroll
  for (int i = 0; i < 8; ++i) { a[i] = lo[i]; a[i + 8] = hi[i]; }
  return a;
}

// B (32x16 f16, K x N): lane l holds column n=l&15; halves 0..7 -> K = off8 + j,
// halves 8..15 -> K = 16 + off8 + (j-8), off8 = (l&16)?8:0. Source stored "n-major" [n][k] (i.e. W row-major
// gives exactly B = W^T; K stored token-major gives B = K^T).
__device__ __forceinline__ v16h load_bt_frag(const _Float16* base, int ld, int k0) {
  int lane = threadIdx.x & 31;
  int n    = lane & 15;
  int off  = (lane & 16) ? 8 : 0;
  const _Float16* p = base + n * ld + k0 + off;
  v8h lo = *(const v8h*)(p);
  v8h hi = *(const v8h*)(p + 16);
  v16h b;
#pragma unroll
  for (int i = 0; i < 8; ++i) { b[i] = lo[i]; b[i + 8] = hi[i]; }
  return b;
}

__device__ __forceinline__ v8f wmma16(v16h a, v16h b, v8f c) {
  return __builtin_amdgcn_wmma_f32_16x16x32_f16(false, a, false, b, (short)0, c,
                                                false, false);
}

// ---------------------------------------------------------------------------
// Prep: f32 -> f16 weights, pre-gathered relative position bias [h][n][m].
// ---------------------------------------------------------------------------
__global__ void prep_kernel(const float* __restrict__ w_qkv,
                            const float* __restrict__ w_skip,
                            const float* __restrict__ w_proj,
                            const int*   __restrict__ rel_idx,
                            const float* __restrict__ rpb,
                            _Float16* __restrict__ wqkv_h,
                            _Float16* __restrict__ wskip_h,
                            _Float16* __restrict__ wproj_h,
                            float*    __restrict__ biasPre) {
  int i = blockIdx.x * 256 + threadIdx.x;
  if (i < 2 * C_DIM * C_DIM) wqkv_h[i] = (_Float16)w_qkv[i];
  if (i < C_DIM * C_DIM) {
    wskip_h[i] = (_Float16)w_skip[i];
    wproj_h[i] = (_Float16)w_proj[i];
  }
  if (i < HEADS * N_TOK * N_TOK) {
    int h  = i / (N_TOK * N_TOK);
    int nm = i % (N_TOK * N_TOK);
    biasPre[i] = rpb[rel_idx[nm] * HEADS + h];
  }
}

// ---------------------------------------------------------------------------
// Fused window-MSA: one workgroup (4 waves) per window.
// ---------------------------------------------------------------------------
__global__ __launch_bounds__(128, 1)
void msa_kernel(const float* __restrict__ x, const float* __restrict__ skip_x,
                const float* __restrict__ mask,
                const _Float16* __restrict__ wqkv,
                const _Float16* __restrict__ wskip,
                const _Float16* __restrict__ wproj,
                const float* __restrict__ biasPre,
                const float* __restrict__ bqkv, const float* __restrict__ bskip,
                const float* __restrict__ bproj, float* __restrict__ out) {
  extern __shared__ char smem[];
  // Region 1 (48 KB): x & skip_x (f16) -> reused for scores/P/out-accum later.
  _Float16* xls  = (_Float16*)smem;              // [64][192]
  _Float16* skls = xls + N_PAD * C_DIM;          // [64][192]
  _Float16* qls  = skls + N_PAD * C_DIM;         // [64][192]
  _Float16* kls  = qls + N_PAD * C_DIM;          // [64][192]
  _Float16* vTl  = kls + N_PAD * C_DIM;          // [192][64] (feature-major)
  float*    s_sc = (float*)smem;                 // [64][64]  (16 KB)
  _Float16* pls  = (_Float16*)(smem + N_PAD * N_PAD * 4);              // [64][64]
  _Float16* ols  = (_Float16*)(smem + N_PAD * N_PAD * 4 + N_PAD * N_PAD * 2); // [64][192]

  const int blk  = blockIdx.x;
  const int tid  = threadIdx.x;
  const int wave = tid >> 5;
  const int lane = tid & 31;
  const int ln   = lane & 15;
  const int ro   = (lane & 16) ? 8 : 0;
  const float scale = 0.17677669529663687f; // 1/sqrt(32)

  // ---- Load x, skip_x into LDS as f16; zero pad rows 49..63 ----
  const float* xg = x      + (size_t)blk * N_TOK * C_DIM;
  const float* sg = skip_x + (size_t)blk * N_TOK * C_DIM;
  for (int i = tid; i < N_PAD * C_DIM; i += 128) {
    int r = i / C_DIM, c = i - r * C_DIM;
    if (r < N_TOK) {
      xls[i]  = (_Float16)xg[(size_t)r * C_DIM + c];
      skls[i] = (_Float16)sg[(size_t)r * C_DIM + c];
    } else {
      xls[i]  = (_Float16)0.f;
      skls[i] = (_Float16)0.f;
    }
  }
  __syncthreads();

  // ---- Stage A: Q = (skip_x @ Wskip^T + b) * scale, per-wave 16-row tile ----
  {
    const _Float16* abase = skls + wave * 16 * C_DIM;
    v16h afr[6];
#pragma unroll
    for (int kt = 0; kt < 6; ++kt) afr[kt] = load_a_frag(abase, C_DIM, kt * 32);
    for (int nt = 0; nt < 12; ++nt) {
      v8f acc = {};
#pragma unroll
      for (int kt = 0; kt < 6; ++kt)
        acc = wmma16(afr[kt], load_bt_frag(wskip + nt * 16 * C_DIM, C_DIM, kt * 32), acc);
      int col = nt * 16 + ln;
      float bb = bskip[col];
#pragma unroll
      for (int r = 0; r < 8; ++r)
        qls[(wave * 16 + ro + r) * C_DIM + col] = (_Float16)((acc[r] + bb) * scale);
    }
  }

  // ---- Stage B: KV = x @ Wqkv^T + b; K token-major, V feature-major ----
  {
    const _Float16* abase = xls + wave * 16 * C_DIM;
    v16h afr[6];
#pragma unroll
    for (int kt = 0; kt < 6; ++kt) afr[kt] = load_a_frag(abase, C_DIM, kt * 32);
    for (int nt = 0; nt < 24; ++nt) {
      v8f acc = {};
#pragma unroll
      for (int kt = 0; kt < 6; ++kt)
        acc = wmma16(afr[kt], load_bt_frag(wqkv + nt * 16 * C_DIM, C_DIM, kt * 32), acc);
      int col = nt * 16 + ln;
      float bb = bqkv[col];
      if (nt < 12) {
#pragma unroll
        for (int r = 0; r < 8; ++r)
          kls[(wave * 16 + ro + r) * C_DIM + col] = (_Float16)(acc[r] + bb);
      } else {
        _Float16* dst = vTl + (col - C_DIM) * N_PAD + wave * 16 + ro;
#pragma unroll
        for (int r = 0; r < 8; ++r) dst[r] = (_Float16)(acc[r] + bb);
      }
    }
  }
  __syncthreads();

  // zero P pad rows (49..63) once
  for (int i = tid; i < (N_PAD - N_TOK) * N_PAD; i += 128)
    pls[(N_TOK + i / N_PAD) * N_PAD + (i % N_PAD)] = (_Float16)0.f;

  const float* mrow = mask + (size_t)(blk & (NWIN - 1)) * N_TOK * N_TOK;

  // ---- Stage C: per-head attention ----
  for (int h = 0; h < HEADS; ++h) {
    const float* bh = biasPre + h * N_TOK * N_TOK;
    // S = Q_h @ K_h^T : 16 tiles, 4 per wave; bias + mask fused into the store
#pragma unroll
    for (int i = 0; i < 4; ++i) {
      int t = wave * 4 + i;
      int mt = t >> 2, nt2 = t & 3;
      v16h a = load_a_frag(qls + mt * 16 * C_DIM, C_DIM, h * HD);
      v16h b = load_bt_frag(kls + nt2 * 16 * C_DIM, C_DIM, h * HD);
      v8f acc = {};
      acc = wmma16(a, b, acc);
      int col = nt2 * 16 + ln;
#pragma unroll
      for (int r = 0; r < 8; ++r) {
        int m = mt * 16 + ro + r;
        float v = acc[r];
        if (m < N_TOK && col < N_TOK)
          v += bh[m * N_TOK + col] + mrow[m * N_TOK + col];
        s_sc[m * N_PAD + col] = v;
      }
    }
    __syncthreads();

    // softmax: 2 threads per row (interleaved columns), pair-reduce via shfl,
    // exp values live in registers (fully unrolled constant-trip loops).
    if (tid < 2 * N_TOK) {
      const int m    = tid >> 1;
      const int half = tid & 1;
      float vals[25];
      float mx = -1e30f;
#pragma unroll
      for (int i = 0; i < 25; ++i) {
        int j = 2 * i + half;
        float v = (j < N_TOK) ? s_sc[m * N_PAD + j] : -1e30f;
        vals[i] = v;
        mx = fmaxf(mx, v);
      }
      mx = fmaxf(mx, __shfl_xor(mx, 1, 32));
      float sum = 0.f;
#pragma unroll
      for (int i = 0; i < 25; ++i) {
        float e = __expf(vals[i] - mx);
        vals[i] = e;
        sum += e;
      }
      // drop the j==49 ghost column contribution of half==1 (exp(-inf-mx)=0
      // anyway since vals was -1e30, expf underflows to 0) and pair-reduce
      sum += __shfl_xor(sum, 1, 32);
      float inv = 1.f / sum;
#pragma unroll
      for (int i = 0; i < 25; ++i) {
        int j = 2 * i + half;
        if (j < N_TOK) pls[m * N_PAD + j] = (_Float16)(vals[i] * inv);
      }
      // zero the pad columns 49..63 of this row
      for (int j = N_TOK + half; j < N_PAD; j += 2)
        pls[m * N_PAD + j] = (_Float16)0.f;
    }
    __syncthreads();

    // O_h = P @ V_h : 8 tiles (4 row-tiles x 2 dim-tiles), 2 per wave, K=64
#pragma unroll
    for (int i = 0; i < 2; ++i) {
      int t = wave * 2 + i;
      int mt = t >> 1, dt = t & 1;
      v8f acc = {};
#pragma unroll
      for (int kt = 0; kt < 2; ++kt) {
        v16h a = load_a_frag(pls + mt * 16 * N_PAD, N_PAD, kt * 32);
        v16h b = load_bt_frag(vTl + (h * HD + dt * 16) * N_PAD, N_PAD, kt * 32);
        acc = wmma16(a, b, acc);
      }
      int col = h * HD + dt * 16 + ln;
#pragma unroll
      for (int r = 0; r < 8; ++r)
        ols[(mt * 16 + ro + r) * C_DIM + col] = (_Float16)acc[r];
    }
    // next head's S-store only happens after every thread passed both barriers
  }
  __syncthreads();

  // ---- Stage D: out = O @ Wproj^T + b (f32 to HBM) ----
  {
    const _Float16* abase = ols + wave * 16 * C_DIM;
    v16h afr[6];
#pragma unroll
    for (int kt = 0; kt < 6; ++kt) afr[kt] = load_a_frag(abase, C_DIM, kt * 32);
    float* og = out + (size_t)blk * N_TOK * C_DIM;
    for (int nt = 0; nt < 12; ++nt) {
      v8f acc = {};
#pragma unroll
      for (int kt = 0; kt < 6; ++kt)
        acc = wmma16(afr[kt], load_bt_frag(wproj + nt * 16 * C_DIM, C_DIM, kt * 32), acc);
      int col = nt * 16 + ln;
      float bb = bproj[col];
#pragma unroll
      for (int r = 0; r < 8; ++r) {
        int m = wave * 16 + ro + r;
        if (m < N_TOK) og[(size_t)m * C_DIM + col] = acc[r] + bb;
      }
    }
  }
}

extern "C" void kernel_launch(void* const* d_in, const int* in_sizes, int n_in,
                              void* d_out, int out_size, void* d_ws, size_t ws_size,
                              hipStream_t stream) {
  const float* x      = (const float*)d_in[0];
  const float* skip_x = (const float*)d_in[1];
  const float* mask   = (const float*)d_in[2];
  const int*   relidx = (const int*)d_in[3];
  const float* rpb    = (const float*)d_in[4];
  const float* w_qkv  = (const float*)d_in[5];
  const float* b_qkv  = (const float*)d_in[6];
  const float* w_skip = (const float*)d_in[7];
  const float* b_skip = (const float*)d_in[8];
  const float* w_proj = (const float*)d_in[9];
  const float* b_proj = (const float*)d_in[10];
  float* out = (float*)d_out;

  // workspace: f16 weights + pre-gathered bias (~353 KB)
  _Float16* wqkv_h  = (_Float16*)d_ws;
  _Float16* wskip_h = wqkv_h + 2 * C_DIM * C_DIM;
  _Float16* wproj_h = wskip_h + C_DIM * C_DIM;
  float*    biasPre = (float*)(wproj_h + C_DIM * C_DIM);

  prep_kernel<<<(2 * C_DIM * C_DIM + 255) / 256, 256, 0, stream>>>(
      w_qkv, w_skip, w_proj, relidx, rpb, wqkv_h, wskip_h, wproj_h, biasPre);

  hipFuncSetAttribute((const void*)msa_kernel,
                      hipFuncAttributeMaxDynamicSharedMemorySize, 122880);
  msa_kernel<<<4096, 128, 122880, stream>>>(x, skip_x, mask, wqkv_h, wskip_h,
                                            wproj_h, biasPre, b_qkv, b_skip,
                                            b_proj, out);
  (void)in_sizes; (void)n_in; (void)out_size; (void)ws_size;
}
